// Reservoir_13185549598993
// MI455X (gfx1250) — matrix operand order repaired and checked by reading, not compiled, not run on hardware
//
#include <hip/hip_runtime.h>
#include <hip/hip_bf16.h>

// Reservoir (echo-state) recurrence for gfx1250:
//   h_t = tanh(W_in x_t + b + W_h h_{t-1})
// Phase 1: U[t] = x_t W_in^T + b  -> written straight into d_out (parallel).
// Phase 2: persistent WMMA kernel, W_h slice pinned in LDS, grid barrier/step.
//   - 64 blocks x 512 threads (16 waves): 2 M-tiles x 2 N-tiles x 4 K-quarters
//   - 2-deep software pipeline of A/B fragments + dual accumulators
//   - U[t] prefetched before the K loop; branch-free tanh.

typedef unsigned short ushort_t;
typedef __attribute__((ext_vector_type(16))) __bf16 v16bf;
typedef __attribute__((ext_vector_type(8)))  float  v8f;

#define RES_H        2048
#define RES_B        32
#define RES_L        1024
#define RES_IN       128
#define GBLOCKS      64
#define COLS_PER_BLK 32
#define WROW         (RES_H + 8)   // padded LDS row stride (ushorts): 4112 B -> 4-bank shift/row
#define KQ_LEN       (RES_H / 4)   // 512: K quarter per wave group
#define KQ_IT        (KQ_LEN / 32) // 16 WMMA iterations per wave per step

__device__ __forceinline__ ushort_t f32_to_bf16_rne(float f) {
  unsigned int u = __float_as_uint(f);
  u += 0x7FFFu + ((u >> 16) & 1u);
  return (ushort_t)(u >> 16);
}

// Branch-free tanh: clamp + exp + rcp (v_exp_f32 / v_rcp_f32, no exec-mask forks).
__device__ __forceinline__ float fast_tanh(float x) {
  x = fminf(fmaxf(x, -15.0f), 15.0f);
  float e = __expf(2.0f * x);
  return (e - 1.0f) * __builtin_amdgcn_rcpf(e + 1.0f);
}

union FragBF { v16bf v; uint4 u[2]; };

// ---------------------------------------------------------------- init -----
__global__ void res_init(const float* __restrict__ h0,
                         ushort_t* __restrict__ hbuf,
                         unsigned* __restrict__ ctr) {
  if (blockIdx.x == 0 && threadIdx.x == 0) *ctr = 0u;
  int i = blockIdx.x * blockDim.x + threadIdx.x;   // 16384 threads
  #pragma unroll
  for (int e = 0; e < 4; ++e) {
    int idx = i + e * 16384;                       // covers RES_B*RES_H = 65536
    hbuf[idx] = f32_to_bf16_rne(h0[idx]);
  }
}

// ------------------------------------------------- W_h f32 -> bf16 ---------
__global__ void res_convw(const float* __restrict__ W, ushort_t* __restrict__ Wb) {
  size_t i = (size_t)blockIdx.x * blockDim.x + threadIdx.x;  // 1M threads x float4
  float4 w = ((const float4*)W)[i];
  unsigned lo = (unsigned)f32_to_bf16_rne(w.x) | ((unsigned)f32_to_bf16_rne(w.y) << 16);
  unsigned hi = (unsigned)f32_to_bf16_rne(w.z) | ((unsigned)f32_to_bf16_rne(w.w) << 16);
  ((uint2*)Wb)[i] = make_uint2(lo, hi);
}

// ----------------------------------- U[t] = x_t W_in^T + b  (into d_out) ---
__global__ void res_u(const float* __restrict__ x, const float* __restrict__ Win,
                      const float* __restrict__ bias, float* __restrict__ out) {
  __shared__ float xs[RES_IN];
  size_t row = blockIdx.x;                      // t*BATCH + b, 32768 blocks
  xs[threadIdx.x] = x[row * RES_IN + threadIdx.x];
  __syncthreads();
  size_t base = row * RES_H;
  #pragma unroll 1
  for (int jj = 0; jj < RES_H / 128; ++jj) {
    int j = threadIdx.x + (jj << 7);
    const float4* wr = (const float4*)(Win + (size_t)j * RES_IN);
    float acc = bias[j];
    #pragma unroll
    for (int k = 0; k < RES_IN / 4; ++k) {
      float4 w = wr[k];
      acc += xs[4*k]*w.x + xs[4*k+1]*w.y + xs[4*k+2]*w.z + xs[4*k+3]*w.w;
    }
    out[base + j] = acc;
  }
}

// Load one 16x32 A fragment and 32x16 B fragment for K offset (in ushorts).
#define LOADFRAG(FA, FB, KOFF)                                   \
  do {                                                           \
    (FA).u[0] = *(const uint4*)(arow + (KOFF));                  \
    (FA).u[1] = *(const uint4*)(arow + (KOFF) + 8);              \
    (FB).u[0] = *(const uint4*)(brow + (KOFF));                  \
    (FB).u[1] = *(const uint4*)(brow + (KOFF) + 8);              \
  } while (0)

#define WMMA_BF16(A, B, C)                                       \
  __builtin_amdgcn_wmma_f32_16x16x32_bf16(false, (A).v, false, (B).v, (short)0, (C), false, false)

// ---------------------------- persistent recurrent kernel ------------------
__global__ void __launch_bounds__(512, 1)
res_step(float* __restrict__ out, const ushort_t* __restrict__ Whb,
         ushort_t* __restrict__ hbuf, unsigned* __restrict__ ctr) {
  extern __shared__ char smem[];
  ushort_t* Wlds = (ushort_t*)smem;                               // 32 x WROW bf16
  float*    red  = (float*)(smem + (size_t)COLS_PER_BLK * WROW * sizeof(ushort_t));

  const int tid  = threadIdx.x;
  const int lane = tid & 31;
  const int wave = tid >> 5;          // 16 waves
  const int jblk = blockIdx.x * COLS_PER_BLK;

  // Stage this block's W_h slice (rows jblk..jblk+31) into padded LDS, once.
  for (int r = wave; r < COLS_PER_BLK; r += 16) {
    const uint4* src = (const uint4*)(Whb + (size_t)(jblk + r) * RES_H);
    uint4*       dst = (uint4*)(Wlds + (size_t)r * WROW);
    for (int c = lane; c < RES_H / 8; c += 32) dst[c] = src[c];
  }
  __syncthreads();

  const int pairID = wave & 3;        // tile pair id
  const int mt = pairID & 1;          // batch tile (M): 0..1
  const int jt = pairID >> 1;         // column tile (N): 0..1
  const int kq = wave >> 2;           // K quarter: 0..3
  const int lr = lane & 15;
  const int hi = lane >> 4;
  const int j  = jblk + jt * 16 + lr; // hidden column owned at finalize (kq==0)

  // B fragment base: column (jt*16+lr) of this block's W slice; contiguous K order (same as A):
  // lane n covers K 0..15, lane n+16 covers K 16..31 (contiguous 32B each).
  const ushort_t* brow = Wlds + (size_t)(jt * 16 + lr) * WROW + hi * 16;
  const int kbase = kq * KQ_LEN;

  for (int t = 0; t < RES_L; ++t) {
    const ushort_t* hprev = hbuf + (size_t)(t & 1) * RES_B * RES_H;
    ushort_t*       hnext = hbuf + (size_t)((t + 1) & 1) * RES_B * RES_H;
    // A fragment: row (mt*16+lr) of h; lane m: K 0..15, lane m+16: K 16..31.
    const ushort_t* arow = hprev + (size_t)(mt * 16 + lr) * RES_H + hi * 16;
    const size_t obase = (size_t)t * RES_B * RES_H;

    // Prefetch U[t] (independent of h) so it lands under the WMMA loop.
    float upre[8];
    if (kq == 0) {
      #pragma unroll
      for (int v = 0; v < 8; ++v)
        upre[v] = out[obase + (size_t)(mt * 16 + hi * 8 + v) * RES_H + j];
    }

    // 2-deep software-pipelined K loop with dual accumulators.
    v8f acc0 = {}, acc1 = {};
    FragBF fa0, fb0, fa1, fb1;
    LOADFRAG(fa0, fb0, kbase);
    #pragma unroll
    for (int p = 0; p < KQ_IT / 2 - 1; ++p) {
      LOADFRAG(fa1, fb1, kbase + (2 * p + 1) * 32);
      acc0 = WMMA_BF16(fa0, fb0, acc0);
      LOADFRAG(fa0, fb0, kbase + (2 * p + 2) * 32);
      acc1 = WMMA_BF16(fa1, fb1, acc1);
    }
    LOADFRAG(fa1, fb1, kbase + (KQ_IT - 1) * 32);
    acc0 = WMMA_BF16(fa0, fb0, acc0);
    acc1 = WMMA_BF16(fa1, fb1, acc1);
    v8f acc = acc0 + acc1;

    // Cross-wave K reduction via LDS (kq 1..3 -> kq 0).
    if (kq != 0) {
      float* rp = red + (((size_t)(kq - 1) * 4 + pairID) * 32 + lane) * 8;
      #pragma unroll
      for (int v = 0; v < 8; ++v) rp[v] = acc[v];
    }
    __syncthreads();
    if (kq == 0) {
      #pragma unroll
      for (int s = 0; s < 3; ++s) {
        const float* rp = red + (((size_t)s * 4 + pairID) * 32 + lane) * 8;
        #pragma unroll
        for (int v = 0; v < 8; ++v) acc[v] += rp[v];
      }
      #pragma unroll
      for (int v = 0; v < 8; ++v) {
        int brow_idx = mt * 16 + hi * 8 + v;     // batch row (D layout: M=v+8*hi)
        size_t oidx = obase + (size_t)brow_idx * RES_H + j;
        float s = fast_tanh(acc[v] + upre[v]);   // out held U[t]; overwrite with h_t
        out[oidx] = s;
        hnext[(size_t)brow_idx * RES_H + j] = f32_to_bf16_rne(s);
      }
    }

    // Grid-wide barrier: monotonic counter, release stores / acquire loads.
    __syncthreads();
    if (tid == 0) {
      __threadfence();
      __hip_atomic_fetch_add(ctr, 1u, __ATOMIC_RELEASE, __HIP_MEMORY_SCOPE_AGENT);
      const unsigned target = (unsigned)GBLOCKS * (unsigned)(t + 1);
      while (__hip_atomic_load(ctr, __ATOMIC_ACQUIRE, __HIP_MEMORY_SCOPE_AGENT) < target)
        __builtin_amdgcn_s_sleep(2);
    }
    __syncthreads();
  }
}

// --------------------------------------------------------------------------
extern "C" void kernel_launch(void* const* d_in, const int* in_sizes, int n_in,
                              void* d_out, int out_size, void* d_ws, size_t ws_size,
                              hipStream_t stream) {
  const float* x    = (const float*)d_in[0];   // (L, B, 128)
  const float* h0   = (const float*)d_in[1];   // (B, H)
  const float* Win  = (const float*)d_in[2];   // (H, 128)
  const float* bias = (const float*)d_in[3];   // (H,)
  const float* Wh   = (const float*)d_in[4];   // (H, H)
  // d_in[5] = washout (0) -> full state sequence returned.
  float* out = (float*)d_out;                  // (L, B, H) f32

  // Workspace: W_h bf16 (8 MB) | h double buffer bf16 (256 KB) | barrier ctr.
  ushort_t* Whb  = (ushort_t*)d_ws;
  ushort_t* hbuf = Whb + (size_t)RES_H * RES_H;
  unsigned* ctr  = (unsigned*)(hbuf + 2 * (size_t)RES_B * RES_H);

  res_init<<<64, 256, 0, stream>>>(h0, hbuf, ctr);
  res_convw<<<(RES_H * RES_H / 4) / 256, 256, 0, stream>>>(Wh, Whb);
  res_u<<<RES_L * RES_B, RES_IN, 0, stream>>>(x, Win, bias, out);

  size_t shmem = (size_t)COLS_PER_BLK * WROW * sizeof(ushort_t)   // 131584 B
               + (size_t)12 * 32 * 8 * sizeof(float);             // 12288 B
  res_step<<<GBLOCKS, 512, shmem, stream>>>(out, Whb, hbuf, ctr);
}
